// MyGPT_12524124635616
// MI455X (gfx1250) — matrix operand order repaired and checked by reading, not compiled, not run on hardware
//
#include <hip/hip_runtime.h>
#include <stdint.h>
#include <stddef.h>

// ---------------- model dims ----------------
#define B_   2
#define T_   2048
#define D_   1024
#define H_   16
#define DQK_ 64
#define DV_  64
#define HD_  4096
#define V_   32000
#define L_   4
#define M_   (B_ * T_)   // 4096 rows of activations
#define BK_  64          // K-chunk staged in LDS per TDM transfer

typedef __bf16 bf16_t;
typedef __attribute__((ext_vector_type(16))) __bf16 v16bf;
typedef __attribute__((ext_vector_type(8)))  __bf16 v8bf;
typedef __attribute__((ext_vector_type(8)))  float  v8f;
typedef __attribute__((ext_vector_type(4)))  unsigned int v4u;
typedef __attribute__((ext_vector_type(8)))  int v8i;
typedef __attribute__((ext_vector_type(4)))  int v4i;

#if __has_builtin(__builtin_amdgcn_tensor_load_to_lds) && \
    __has_builtin(__builtin_amdgcn_s_wait_tensorcnt)
#define HAVE_TDM 1
#else
#define HAVE_TDM 0
#endif

// A-operand fragment (16-bit A 16x32): lane half hi=0 -> K {0..15},
// hi=1 -> K {16..31}. Caller passes p = rowbase + kk + hi*16.
static __device__ __forceinline__ v16bf load_fragA(const bf16_t* p) {
  v8bf lo = *(const v8bf*)(p);        // K +0..7   (relative)
  v8bf hi = *(const v8bf*)(p + 8);    // K +8..15  (relative)
  return __builtin_shufflevector(lo, hi, 0,1,2,3,4,5,6,7,8,9,10,11,12,13,14,15);
}

static __device__ __forceinline__ v8f wmma_bf16(v16bf a, v16bf b, v8f c) {
  return __builtin_amdgcn_wmma_f32_16x16x32_bf16(false, a, false, b, (short)0, c,
                                                 false, false);
}

#if HAVE_TDM
// Issue a TDM load of a 64-row x BK_-col bf16 tile of W[N,K] (row-major,
// row stride K elements) into LDS at byte offset lds_off.
static __device__ __forceinline__ void tdm_load_tile(unsigned lds_off,
                                                     const bf16_t* gaddr,
                                                     int K, int N) {
  unsigned long long ga = (unsigned long long)(uintptr_t)gaddr;
  v4u g0;
  g0[0] = 1u;                                              // count=1, user mode
  g0[1] = lds_off;                                         // lds_addr [63:32]
  g0[2] = (unsigned)(ga & 0xffffffffu);                    // global_addr lo
  g0[3] = (unsigned)((ga >> 32) & 0x01ffffffu) | (2u << 30); // ga hi | type=2
  v8i g1;
  g1[0] = 0x00010000;                                      // data_size=1 (2B)
  g1[1] = (int)(((unsigned)K & 0xffffu) << 16);            // tensor_dim0 lo
  g1[2] = (int)((((unsigned)K >> 16) & 0xffffu) |
                (((unsigned)N & 0xffffu) << 16));          // td0 hi | td1 lo
  g1[3] = (int)((((unsigned)N >> 16) & 0xffffu) |
                ((unsigned)BK_ << 16));                    // td1 hi | tile_dim0
  g1[4] = 64;                                              // tile_dim1=64, td2=0
  g1[5] = K;                                               // dim0_stride lo
  g1[6] = 0;                                               // stride hi | st1 lo
  g1[7] = 0;
  v4i z = {0, 0, 0, 0};
#if __clang_major__ >= 23
  v8i z8 = {0, 0, 0, 0, 0, 0, 0, 0};
  __builtin_amdgcn_tensor_load_to_lds(g0, g1, z, z, z8, 0);
#else
  __builtin_amdgcn_tensor_load_to_lds(g0, g1, z, z, 0);
#endif
}
#endif

// ---------------- elementwise / setup kernels ----------------
__global__ void k_cvt_bf16(const float* __restrict__ in, bf16_t* __restrict__ out, long n) {
  long i = (long)blockIdx.x * 256 + threadIdx.x;
  if (i < n) out[i] = (bf16_t)in[i];
}

// w_o [L][K=1024][N=1024] (in,out) -> woT [L][N][K] bf16
__global__ void k_tr_wo(const float* __restrict__ in, bf16_t* __restrict__ out) {
  long i = (long)blockIdx.x * 256 + threadIdx.x;
  if (i >= (long)L_ * D_ * D_) return;
  int l = (int)(i >> 20);
  int rem = (int)(i & ((1 << 20) - 1));
  int n = rem >> 10, kk = rem & 1023;
  out[i] = (bf16_t)in[((size_t)l << 20) + (size_t)kk * D_ + n];
}

// v [B,T,H,DV] bf16 -> vT [B,H,DV,T] bf16
__global__ void k_tr_v(const bf16_t* __restrict__ in, bf16_t* __restrict__ out) {
  long i = (long)blockIdx.x * 256 + threadIdx.x;
  if (i >= (long)B_ * H_ * DV_ * T_) return;
  int t  = (int)(i & (T_ - 1));
  int dv = (int)((i >> 11) & (DV_ - 1));
  int h  = (int)((i >> 17) & (H_ - 1));
  int b  = (int)((i >> 21) & 1);
  out[i] = in[((size_t)(b * T_ + t) * H_ + h) * DV_ + dv];
}

// x[b,t,d] = emb[shifted_tok] + sin(t / LEN_MAX^((j-k)/D) + (pi/2)*k)
__global__ void k_embed(const int* __restrict__ tokens, const float* __restrict__ emb,
                        float* __restrict__ x) {
  long i = (long)blockIdx.x * 256 + threadIdx.x;
  if (i >= (long)B_ * T_ * D_) return;
  int d = (int)(i & (D_ - 1));
  long bt = i >> 10;
  int t = (int)(bt & (T_ - 1));
  int b = (int)(bt >> 11);
  int tok = (t == 0) ? 0 : tokens[(size_t)b * T_ + t - 1];
  float kpar = (float)(d & 1);
  float expo = ((float)d - kpar) * (1.0f / (float)D_);
  float freq = __powf(100000.0f, expo);
  float pos  = __sinf((float)t / freq + 1.5707963267948966f * kpar);
  x[i] = emb[(size_t)tok * D_ + d] + pos;
}

// ---------------- LayerNorm: one block (256 thr) per row ----------------
__global__ void __launch_bounds__(256) k_layernorm(
    const float* __restrict__ x, const float* __restrict__ g,
    const float* __restrict__ bta, bf16_t* __restrict__ out) {
  __shared__ float sh[16];
  int row = blockIdx.x, tid = threadIdx.x;
  float4 v = ((const float4*)(x + (size_t)row * D_))[tid];
  float s  = v.x + v.y + v.z + v.w;
  float qq = v.x * v.x + v.y * v.y + v.z * v.z + v.w * v.w;
#pragma unroll
  for (int m = 16; m >= 1; m >>= 1) {
    s  += __shfl_xor(s, m, 32);
    qq += __shfl_xor(qq, m, 32);
  }
  if ((tid & 31) == 0) { sh[tid >> 5] = s; sh[8 + (tid >> 5)] = qq; }
  __syncthreads();
  float ts = 0.f, tq = 0.f;
#pragma unroll
  for (int i = 0; i < 8; ++i) { ts += sh[i]; tq += sh[8 + i]; }
  float mean = ts * (1.0f / (float)D_);
  float rstd = rsqrtf(tq * (1.0f / (float)D_) - mean * mean + 1e-5f);
  int c = tid * 4;
  union { bf16_t h[4]; uint2 u; } r;
  r.h[0] = (bf16_t)((v.x - mean) * rstd * g[c + 0] + bta[c + 0]);
  r.h[1] = (bf16_t)((v.y - mean) * rstd * g[c + 1] + bta[c + 1]);
  r.h[2] = (bf16_t)((v.z - mean) * rstd * g[c + 2] + bta[c + 2]);
  r.h[3] = (bf16_t)((v.w - mean) * rstd * g[c + 3] + bta[c + 3]);
  *(uint2*)(out + (size_t)row * D_ + c) = r.u;
}

// ---------------- WMMA GEMM with TDM-staged B tiles ----------------
// C[M,N] = A[M,K] * W[N,K]^T. Block = 8 waves, each wave 32x64 output;
// block output = 256x64. B tile (64 x BK_) double-buffered in LDS via the
// Tensor Data Mover (wave 0 issues, TENSORcnt + barrier synchronize).
__global__ void __launch_bounds__(256) k_gemm(
    const bf16_t* __restrict__ A, const bf16_t* __restrict__ W,
    const float* __restrict__ bias, const float* resid,
    float* outF, bf16_t* outB, int N, int K, int relu) {
  __shared__ bf16_t bsm[2][64 * BK_];   // 2 x 8KB
  int tid = threadIdx.x;
  int wave = tid >> 5, lane = tid & 31;
  int ntiles = N >> 6;
  int nt  = blockIdx.x % ntiles;
  int mtb = blockIdx.x / ntiles;        // 256-row block
  int l16 = lane & 15, hi = lane >> 4;

  int rbase = mtb * 256 + wave * 32;
  const bf16_t* ap0 = A + (size_t)(rbase + l16) * K + hi * 16;
  const bf16_t* ap1 = ap0 + (size_t)16 * K;
  const bf16_t* wtile = W + (size_t)(nt * 64) * K;   // 64 rows of W

  unsigned lds0 = (unsigned)(uintptr_t)(&bsm[0][0]);
  unsigned lds1 = (unsigned)(uintptr_t)(&bsm[1][0]);
  int nchunks = K / BK_;

#if HAVE_TDM
  if (wave == 0) {
    tdm_load_tile(lds0, wtile, K, N);
    __builtin_amdgcn_s_wait_tensorcnt(0);
  }
  __syncthreads();
#else
  // cooperative fallback: 256 threads copy the 64xBK_ tile (32B each)
  {
    int r = tid >> 2, c4 = (tid & 3) * 16;
    *(v16bf*)(&bsm[0][r * BK_ + c4]) = *(const v16bf*)(wtile + (size_t)r * K + c4);
  }
  __syncthreads();
#endif

  v8f accLo[4] = {v8f{}, v8f{}, v8f{}, v8f{}};
  v8f accHi[4] = {v8f{}, v8f{}, v8f{}, v8f{}};

  for (int c = 0; c < nchunks; ++c) {
#if HAVE_TDM
    if (wave == 0 && c + 1 < nchunks)
      tdm_load_tile(((c + 1) & 1) ? lds1 : lds0,
                    wtile + (size_t)(c + 1) * BK_, K, N);
#endif
    const bf16_t* bs = &bsm[c & 1][0];
    const bf16_t* a0 = ap0 + (size_t)c * BK_;
    const bf16_t* a1 = ap1 + (size_t)c * BK_;
    __builtin_prefetch(a0 + BK_, 0, 3);
#pragma unroll
    for (int kk2 = 0; kk2 < BK_; kk2 += 32) {
      v16bf af0 = load_fragA(a0 + kk2);
      v16bf af1 = load_fragA(a1 + kk2);
      // materialize all four B fragments in distinct registers so all 8
      // ds_load_b128 are in flight before the 8-WMMA chain issues
      v16bf bfr0 = *(const v16bf*)(bs + (size_t)(0 * 16 + l16) * BK_ + kk2 + hi * 16);
      v16bf bfr1 = *(const v16bf*)(bs + (size_t)(1 * 16 + l16) * BK_ + kk2 + hi * 16);
      v16bf bfr2 = *(const v16bf*)(bs + (size_t)(2 * 16 + l16) * BK_ + kk2 + hi * 16);
      v16bf bfr3 = *(const v16bf*)(bs + (size_t)(3 * 16 + l16) * BK_ + kk2 + hi * 16);
      accLo[0] = wmma_bf16(af0, bfr0, accLo[0]);
      accHi[0] = wmma_bf16(af1, bfr0, accHi[0]);
      accLo[1] = wmma_bf16(af0, bfr1, accLo[1]);
      accHi[1] = wmma_bf16(af1, bfr1, accHi[1]);
      accLo[2] = wmma_bf16(af0, bfr2, accLo[2]);
      accHi[2] = wmma_bf16(af1, bfr2, accHi[2]);
      accLo[3] = wmma_bf16(af0, bfr3, accLo[3]);
      accHi[3] = wmma_bf16(af1, bfr3, accHi[3]);
    }
#if HAVE_TDM
    if (wave == 0) __builtin_amdgcn_s_wait_tensorcnt(0);
    __syncthreads();
#else
    __syncthreads();
    if (c + 1 < nchunks) {
      int r = tid >> 2, c4 = (tid & 3) * 16;
      *(v16bf*)(&bsm[(c + 1) & 1][r * BK_ + c4]) =
          *(const v16bf*)(wtile + (size_t)r * K + (size_t)(c + 1) * BK_ + c4);
      __syncthreads();
    }
#endif
  }

#pragma unroll
  for (int half = 0; half < 2; ++half) {
    int m0 = rbase + half * 16 + hi * 8;
    v8f* acc = half ? accHi : accLo;
#pragma unroll
    for (int j = 0; j < 4; ++j) {
      int col = nt * 64 + j * 16 + l16;
      float bv = bias ? bias[col] : 0.0f;
#pragma unroll
      for (int i = 0; i < 8; ++i) {
        int m = m0 + i;
        float v = acc[j][i] + bv;
        if (relu) v = fmaxf(v, 0.0f);
        if (resid) v += resid[(size_t)m * N + col];
        if (outF) outF[(size_t)m * N + col] = v;
        if (outB) outB[(size_t)m * N + col] = (bf16_t)v;
      }
    }
  }
}

// ---------------- flash attention (transposed form), one wave per 16 t-rows ----
// S^T = K*Q^T  (C layout: rows=s, cols=t -> per-t stats are per-lane)
// O^T = V^T*P^T, P^T built from score tiles with a single xor-16 shuffle.
__global__ void __launch_bounds__(256) k_attn(
    const bf16_t* __restrict__ q, const bf16_t* __restrict__ kmat,
    const bf16_t* __restrict__ vT, bf16_t* __restrict__ y) {
  int lane = threadIdx.x & 31;
  int wid  = blockIdx.x * 8 + (threadIdx.x >> 5);
  int tt = wid & (T_ / 16 - 1);        // t-tile
  int bh = wid >> 7;
  int b = bh >> 4, h = bh & (H_ - 1);
  int l16 = lane & 15, hi = lane >> 4;
  int t_col = tt * 16 + l16;           // this lane's t (column of S^T)

  const bf16_t* qrow = q + ((size_t)(b * T_ + t_col) * H_ + h) * DQK_ + hi * 16;
  v16bf qf0 = *(const v16bf*)(qrow);        // dqk 0..31
  v16bf qf1 = *(const v16bf*)(qrow + 32);   // dqk 32..63

  const size_t kstride = (size_t)H_ * DQK_;
  const bf16_t* kb0 = kmat + (size_t)b * T_ * kstride + (size_t)h * DQK_;
  const bf16_t* vtb = vT + ((size_t)(b * H_ + h) * DV_) * T_;

  float mcur = -1e30f, lsum = 0.0f;
  v8f o[4] = {v8f{}, v8f{}, v8f{}, v8f{}};

  int smax = tt * 16 + 15;
  for (int sb = 0; sb <= smax; sb += 32) {
    const bf16_t* kr0 = kb0 + (size_t)(sb + l16) * kstride + hi * 16;
    const bf16_t* kr1 = kr0 + 16 * kstride;
    v8f s0 = {}, s1 = {};
    s0 = wmma_bf16(load_fragA(kr0),      qf0, s0);
    s0 = wmma_bf16(load_fragA(kr0 + 32), qf1, s0);
    s1 = wmma_bf16(load_fragA(kr1),      qf0, s1);
    s1 = wmma_bf16(load_fragA(kr1 + 32), qf1, s1);

    float cmax = -1e30f;
#pragma unroll
    for (int i = 0; i < 8; ++i) {
      int sr = sb + hi * 8 + i;
      float a0 = s0[i] * 0.125f;                // 1/sqrt(64)
      float a1 = s1[i] * 0.125f;
      if (sr > t_col) a0 = -1e30f;              // causal mask
      if (sr + 16 > t_col) a1 = -1e30f;
      s0[i] = a0; s1[i] = a1;
      cmax = fmaxf(cmax, fmaxf(a0, a1));
    }
    cmax = fmaxf(cmax, __shfl_xor(cmax, 16, 32));
    float mnew = fmaxf(mcur, cmax);
    float corr = __expf(mcur - mnew);
    float csum = 0.0f;
    v16bf pf;
#pragma unroll
    for (int i = 0; i < 8; ++i) {
      float p0 = __expf(s0[i] - mnew);
      float p1 = __expf(s1[i] - mnew);
      csum += p0 + p1;
      float e0 = __shfl_xor(p0, 16, 32);
      float e1 = __shfl_xor(p1, 16, 32);
      float lo = hi ? e1 : p0;
      float hh = hi ? p1 : e0;
      pf[i]     = (bf16_t)lo;
      pf[8 + i] = (bf16_t)hh;
    }
    csum += __shfl_xor(csum, 16, 32);
    lsum = lsum * corr + csum;
    mcur = mnew;
#pragma unroll
    for (int j = 0; j < 4; ++j) {
      o[j] *= corr;
      const bf16_t* vr = vtb + (size_t)(j * 16 + l16) * T_ + sb + hi * 16;
      o[j] = wmma_bf16(load_fragA(vr), pf, o[j]);
    }
  }

  float inv = 1.0f / lsum;
#pragma unroll
  for (int j = 0; j < 4; ++j)
#pragma unroll
    for (int i = 0; i < 8; ++i) {
      int dv = j * 16 + hi * 8 + i;
      y[((size_t)(b * T_ + t_col)) * D_ + h * DV_ + dv] = (bf16_t)(o[j][i] * inv);
    }
}

// ---------------- host orchestration ----------------
static inline int cdiv_i(long n, int b) { return (int)((n + b - 1) / b); }

extern "C" void kernel_launch(void* const* d_in, const int* in_sizes, int n_in,
                              void* d_out, int out_size, void* d_ws, size_t ws_size,
                              hipStream_t stream) {
  (void)in_sizes; (void)n_in; (void)out_size; (void)ws_size;
  const int*   tokens = (const int*)d_in[0];
  const float* emb    = (const float*)d_in[1];
  const float* ln1_w  = (const float*)d_in[2];
  const float* ln1_b  = (const float*)d_in[3];
  const float* w_q    = (const float*)d_in[4];
  const float* w_k    = (const float*)d_in[5];
  const float* w_v    = (const float*)d_in[6];
  const float* w_o    = (const float*)d_in[7];
  const float* ln2_w  = (const float*)d_in[8];
  const float* ln2_b  = (const float*)d_in[9];
  const float* w1     = (const float*)d_in[10];
  const float* b1     = (const float*)d_in[11];
  const float* w2     = (const float*)d_in[12];
  const float* b2     = (const float*)d_in[13];
  const float* ro_w   = (const float*)d_in[14];
  const float* ro_b   = (const float*)d_in[15];
  float* out = (float*)d_out;

  char* ws = (char*)d_ws;
  size_t off = 0;
  auto alloc = [&](size_t bytes) -> void* {
    void* p = ws + off;
    off += (bytes + 255) & ~(size_t)255;
    return p;
  };
  float*  x   = (float*) alloc((size_t)M_ * D_ * 4);
  bf16_t* xn  = (bf16_t*)alloc((size_t)M_ * D_ * 2);
  bf16_t* xb  = (bf16_t*)alloc((size_t)M_ * D_ * 2);
  bf16_t* qb  = (bf16_t*)alloc((size_t)M_ * D_ * 2);
  bf16_t* kb  = (bf16_t*)alloc((size_t)M_ * D_ * 2);
  bf16_t* vb  = (bf16_t*)alloc((size_t)M_ * D_ * 2);
  bf16_t* vTb = (bf16_t*)alloc((size_t)M_ * D_ * 2);
  bf16_t* yb  = (bf16_t*)alloc((size_t)M_ * D_ * 2);
  bf16_t* hb  = (bf16_t*)alloc((size_t)M_ * HD_ * 2);
  bf16_t* wqB  = (bf16_t*)alloc((size_t)L_ * D_ * D_ * 2);
  bf16_t* wkB  = (bf16_t*)alloc((size_t)L_ * D_ * D_ * 2);
  bf16_t* wvB  = (bf16_t*)alloc((size_t)L_ * D_ * D_ * 2);
  bf16_t* woTB = (bf16_t*)alloc((size_t)L_ * D_ * D_ * 2);
  bf16_t* w1B  = (bf16_t*)alloc((size_t)L_ * HD_ * D_ * 2);
  bf16_t* w2B  = (bf16_t*)alloc((size_t)L_ * D_ * HD_ * 2);
  bf16_t* roB  = (bf16_t*)alloc((size_t)V_ * D_ * 2);

  long nw = (long)L_ * D_ * D_;
  k_cvt_bf16<<<cdiv_i(nw, 256), 256, 0, stream>>>(w_q, wqB, nw);
  k_cvt_bf16<<<cdiv_i(nw, 256), 256, 0, stream>>>(w_k, wkB, nw);
  k_cvt_bf16<<<cdiv_i(nw, 256), 256, 0, stream>>>(w_v, wvB, nw);
  k_tr_wo  <<<cdiv_i(nw, 256), 256, 0, stream>>>(w_o, woTB);
  long nm = (long)L_ * HD_ * D_;
  k_cvt_bf16<<<cdiv_i(nm, 256), 256, 0, stream>>>(w1, w1B, nm);
  k_cvt_bf16<<<cdiv_i(nm, 256), 256, 0, stream>>>(w2, w2B, nm);
  long nr = (long)V_ * D_;
  k_cvt_bf16<<<cdiv_i(nr, 256), 256, 0, stream>>>(ro_w, roB, nr);

  k_embed<<<cdiv_i((long)M_ * D_, 256), 256, 0, stream>>>(tokens, emb, x);

  const int gD  = (M_ / 256) * (D_ / 64);    // 256 blocks (N=1024)
  const int gHD = (M_ / 256) * (HD_ / 64);   // 1024 blocks (N=4096)
  const int gV  = (M_ / 256) * (V_ / 64);    // 8000 blocks (N=32000)
  const int gAT = (B_ * H_ * (T_ / 16)) / 8; // 512 blocks

  for (int l = 0; l < L_; ++l) {
    const bf16_t* wq = wqB + (size_t)l * D_ * D_;
    const bf16_t* wk = wkB + (size_t)l * D_ * D_;
    const bf16_t* wv = wvB + (size_t)l * D_ * D_;
    const bf16_t* wo = woTB + (size_t)l * D_ * D_;
    const bf16_t* wm1 = w1B + (size_t)l * HD_ * D_;
    const bf16_t* wm2 = w2B + (size_t)l * D_ * HD_;

    k_layernorm<<<M_, 256, 0, stream>>>(x, ln1_w + l * D_, ln1_b + l * D_, xn);
    k_gemm<<<gD, 256, 0, stream>>>(xn, wq, nullptr, nullptr, nullptr, qb, D_, D_, 0);
    k_gemm<<<gD, 256, 0, stream>>>(xn, wk, nullptr, nullptr, nullptr, kb, D_, D_, 0);
    k_gemm<<<gD, 256, 0, stream>>>(xn, wv, nullptr, nullptr, nullptr, vb, D_, D_, 0);
    k_tr_v<<<cdiv_i((long)M_ * D_, 256), 256, 0, stream>>>(vb, vTb);
    k_attn<<<gAT, 256, 0, stream>>>(qb, kb, vTb, yb);
    k_gemm<<<gD, 256, 0, stream>>>(yb, wo, nullptr, x, x, nullptr, D_, D_, 0);

    k_layernorm<<<M_, 256, 0, stream>>>(x, ln2_w + l * D_, ln2_b + l * D_, xn);
    k_gemm<<<gHD, 256, 0, stream>>>(xn, wm1, b1 + (size_t)l * HD_, nullptr,
                                    nullptr, hb, HD_, D_, 1);
    k_gemm<<<gD, 256, 0, stream>>>(hb, wm2, b2 + (size_t)l * D_, x,
                                   x, (l == L_ - 1) ? xb : nullptr, D_, HD_, 0);
  }

  k_gemm<<<gV, 256, 0, stream>>>(xb, roB, ro_b, nullptr, out, nullptr, V_, D_, 0);
}
